// GatEnconder_tree_gru_60971355734173
// MI455X (gfx1250) — compile-verified, matrix operand order checked by reading, not run on hardware
//
#include <hip/hip_runtime.h>

typedef _Float16 half_t;
typedef __attribute__((ext_vector_type(8)))  _Float16 v8h;
typedef __attribute__((ext_vector_type(16))) _Float16 v16h;
typedef __attribute__((ext_vector_type(8)))  float    v8f;

// ---------------------------------------------------------------- utilities

__global__ void k_f32_to_f16(const float* __restrict__ in, half_t* __restrict__ out, int n) {
  int i = blockIdx.x * blockDim.x + threadIdx.x;
  if (i < n) out[i] = (half_t)in[i];
}

// Ordered-uint encoding for float atomic max (monotonic for all finite floats)
__device__ __forceinline__ unsigned enc_omax(float f) {
  unsigned b = __float_as_uint(f);
  return (b & 0x80000000u) ? ~b : (b | 0x80000000u);
}
__device__ __forceinline__ float dec_omax(unsigned k) {
  if (k == 0u) return 0.0f;  // no in-edges -> matches isfinite fixup in reference
  unsigned b = (k & 0x80000000u) ? (k ^ 0x80000000u) : ~k;
  return __uint_as_float(b);
}

__device__ __forceinline__ v16h pack16(v8h lo, v8h hi) {
  v16h r;
#pragma unroll
  for (int i = 0; i < 8; ++i) { r[i] = lo[i]; r[i + 8] = hi[i]; }
  return r;
}

// ------------------------------------------------------- generic WMMA GEMM
// C[M,Nc] = A[M,K](f16 row-major) x W[Nc,K](f16 row-major)^T  (+bias[Nc])
// One wave computes a 32x32 block (2x2 WMMA tiles): 8 b128 loads feed 4
// WMMAs per k-step of 32 -> 2 loads/WMMA, 16 FLOP/byte at the L2 interface.
// Requires: K % 32 == 0, M % 32 == 0, Nc % 32 == 0 (true for every call here).
__global__ __launch_bounds__(256)
void k_wmma_gemm_nt(const half_t* __restrict__ A, const half_t* __restrict__ Bw,
                    float* __restrict__ C, const float* __restrict__ bias,
                    int M, int Nc, int K) {
  int wave   = blockIdx.x * (blockDim.x >> 5) + (threadIdx.x >> 5);
  int tilesN = Nc >> 5;
  int tiles  = (M >> 5) * tilesN;
  if (wave >= tiles) return;              // wave-uniform: EXEC stays all-1s
  int lane = threadIdx.x & 31;
  int r    = lane & 15;                    // M (for A) / N (for B) index in tile
  int hf   = lane >> 4;                    // lane-half selects K sub-chunk
  int tm = wave / tilesN, tn = wave % tilesN;

  const half_t* a0 = A  + (size_t)(tm * 32 + r) * K + hf * 8;
  const half_t* a1 = a0 + (size_t)16 * K;
  const half_t* b0 = Bw + (size_t)(tn * 32 + r) * K + hf * 8;
  const half_t* b1 = b0 + (size_t)16 * K;

  v8f c00 = {}, c01 = {}, c10 = {}, c11 = {};
#pragma unroll 2
  for (int k0 = 0; k0 < K; k0 += 32) {
    v16h av0 = pack16(*(const v8h*)(a0 + k0), *(const v8h*)(a0 + k0 + 16));
    v16h av1 = pack16(*(const v8h*)(a1 + k0), *(const v8h*)(a1 + k0 + 16));
    v16h bv0 = pack16(*(const v8h*)(b0 + k0), *(const v8h*)(b0 + k0 + 16));
    v16h bv1 = pack16(*(const v8h*)(b1 + k0), *(const v8h*)(b1 + k0 + 16));
    c00 = __builtin_amdgcn_wmma_f32_16x16x32_f16(false, av0, false, bv0, (short)0, c00, false, false);
    c01 = __builtin_amdgcn_wmma_f32_16x16x32_f16(false, av0, false, bv1, (short)0, c01, false, false);
    c10 = __builtin_amdgcn_wmma_f32_16x16x32_f16(false, av1, false, bv0, (short)0, c10, false, false);
    c11 = __builtin_amdgcn_wmma_f32_16x16x32_f16(false, av1, false, bv1, (short)0, c11, false, false);
  }

  int col0 = tn * 32 + r, col1 = col0 + 16;
  float bv0s = bias ? bias[col0] : 0.0f;
  float bv1s = bias ? bias[col1] : 0.0f;
  float* row0 = C + (size_t)(tm * 32 + 8 * hf) * Nc;        // tiles c00/c01
  float* row1 = row0 + (size_t)16 * Nc;                      // tiles c10/c11
#pragma unroll
  for (int i = 0; i < 8; ++i) {
    row0[(size_t)i * Nc + col0] = c00[i] + bv0s;
    row0[(size_t)i * Nc + col1] = c01[i] + bv1s;
    row1[(size_t)i * Nc + col0] = c10[i] + bv0s;
    row1[(size_t)i * Nc + col1] = c11[i] + bv1s;
  }
}

// -------------------------------------------------------------- GNN kernels

// h0 = emb[wid]; accumulate per-graph sums and counts. One block per node.
__global__ void k_embed(const int* __restrict__ wid, const int* __restrict__ n2g,
                        const float* __restrict__ emb, half_t* __restrict__ h0h,
                        float* __restrict__ mean1, float* __restrict__ cnt) {
  int n = blockIdx.x, h = threadIdx.x;
  int w = wid[n], g = n2g[n];
  float v = emb[(size_t)w * 256 + h];
  h0h[(size_t)n * 256 + h] = (half_t)v;
  atomicAdd(&mean1[(size_t)g * 256 + h], v);
  if (h == 0) atomicAdd(&cnt[g], 1.0f);
}

__global__ void k_div_mean1(float* __restrict__ mean1, const float* __restrict__ cnt) {
  int b = blockIdx.x, h = threadIdx.x;
  mean1[(size_t)b * 256 + h] /= fmaxf(cnt[b], 1.0f);
}

// ss[n] = z[n]·a_s, sd[n] = z[n]·a_d. One block (256 thr) per node.
__global__ void k_node_scores(const float* __restrict__ z, const float* __restrict__ as,
                              const float* __restrict__ ad, float* __restrict__ ss,
                              float* __restrict__ sd) {
  __shared__ float r1[256], r2[256];
  int n = blockIdx.x, h = threadIdx.x;
  float zv = z[(size_t)n * 256 + h];
  r1[h] = zv * as[h];
  r2[h] = zv * ad[h];
  __syncthreads();
  for (int off = 128; off > 0; off >>= 1) {
    if (h < off) { r1[h] += r1[h + off]; r2[h] += r2[h + off]; }
    __syncthreads();
  }
  if (h == 0) { ss[n] = r1[0]; sd[n] = r2[0]; }
}

__device__ __forceinline__ float leaky(float x) { return x > 0.0f ? x : 0.01f * x; }

__global__ void k_edge_max(const int* __restrict__ src, const int* __restrict__ dst,
                           const float* __restrict__ ss, const float* __restrict__ sd,
                           unsigned* __restrict__ mU, int E) {
  int k = blockIdx.x * blockDim.x + threadIdx.x;
  if (k >= E) return;
  float e = leaky(ss[src[k]] + sd[dst[k]]);
  atomicMax(&mU[dst[k]], enc_omax(e));
}

__global__ void k_edge_exp(const int* __restrict__ src, const int* __restrict__ dst,
                           const float* __restrict__ ss, const float* __restrict__ sd,
                           const unsigned* __restrict__ mU, float* __restrict__ exb,
                           float* __restrict__ ssum, int E) {
  int k = blockIdx.x * blockDim.x + threadIdx.x;
  if (k >= E) return;
  int d = dst[k];
  float e = leaky(ss[src[k]] + sd[d]);
  float x = expf(e - dec_omax(mU[d]));
  exb[k] = x;
  atomicAdd(&ssum[d], x);
}

// hagg[dst] += alpha * z[src]. One block (256 thr) per edge.
__global__ void k_edge_scatter(const int* __restrict__ src, const int* __restrict__ dst,
                               const float* __restrict__ exb, const float* __restrict__ ssum,
                               const float* __restrict__ z, float* __restrict__ hagg) {
  int k = blockIdx.x, h = threadIdx.x;
  int d = dst[k], s = src[k];
  float sv = ssum[d];
  float alpha = exb[k] / (sv == 0.0f ? 1.0f : sv);
  atomicAdd(&hagg[(size_t)d * 256 + h], alpha * z[(size_t)s * 256 + h]);
}

// Column sums of relu(hagg) for BatchNorm. Row-chunked to cut atomic pressure.
__global__ void k_bn_stats(const float* __restrict__ hagg, float* __restrict__ csum,
                           float* __restrict__ csq, int Nrows) {
  int h = threadIdx.x;
  int r0 = blockIdx.x * 256;
  int r1 = r0 + 256 < Nrows ? r0 + 256 : Nrows;
  float s = 0.0f, q = 0.0f;
  for (int r = r0; r < r1; ++r) {
    float v = fmaxf(hagg[(size_t)r * 256 + h], 0.0f);
    s += v; q += v * v;
  }
  atomicAdd(&csum[h], s);
  atomicAdd(&csq[h], q);
}

__global__ void k_bn_norm(const float* __restrict__ hagg, const float* __restrict__ csum,
                          const float* __restrict__ csq, const float* __restrict__ gamma,
                          const float* __restrict__ beta, half_t* __restrict__ catH,
                          int layer, float invN) {
  int n = blockIdx.x, h = threadIdx.x;
  float mu  = csum[h] * invN;
  float var = csq[h] * invN - mu * mu;
  float hr  = fmaxf(hagg[(size_t)n * 256 + h], 0.0f);
  float v   = (hr - mu) * rsqrtf(var + 1e-5f) * gamma[h] + beta[h];
  catH[(size_t)n * 1024 + layer * 256 + h] = (half_t)v;
}

__global__ void k_mean2_scatter(const int* __restrict__ n2g, const float* __restrict__ newh,
                                float* __restrict__ mean2) {
  int n = blockIdx.x, h = threadIdx.x;
  atomicAdd(&mean2[(size_t)n2g[n] * 256 + h], newh[(size_t)n * 256 + h]);
}

// x[b,0,:]=mean1, x[b,1,:]=mean2/cnt  (f16 for GRU GEMMs)
__global__ void k_build_x(const float* __restrict__ mean1, const float* __restrict__ mean2,
                          const float* __restrict__ cnt, half_t* __restrict__ xh) {
  int b = blockIdx.x, h = threadIdx.x;
  float c = fmaxf(cnt[b], 1.0f);
  xh[(size_t)(b * 2 + 0) * 256 + h] = (half_t)mean1[(size_t)b * 256 + h];
  xh[(size_t)(b * 2 + 1) * 256 + h] = (half_t)(mean2[(size_t)b * 256 + h] / c);
}

// -------------------------------------------------------------- GRU kernels

// One block per batch row. gh_ld==0 -> gh is a broadcast [768] bias vector.
__global__ void k_gru_cell(const float* __restrict__ gi, int gi_ld,
                           const float* __restrict__ gh, int gh_ld,
                           const float* __restrict__ hprev,
                           float* __restrict__ hout, half_t* __restrict__ hout16,
                           half_t* __restrict__ yout, int y_ld) {
  int b = blockIdx.x, h = threadIdx.x;
  const float* gib = gi + (size_t)b * gi_ld;
  const float* ghb = gh + (size_t)b * gh_ld;
  float ir = gib[h], iz = gib[256 + h], inn = gib[512 + h];
  float hr = ghb[h], hz = ghb[256 + h], hn  = ghb[512 + h];
  float hp = hprev ? hprev[(size_t)b * 256 + h] : 0.0f;
  float r  = 1.0f / (1.0f + expf(-(ir + hr)));
  float zg = 1.0f / (1.0f + expf(-(iz + hz)));
  float nn = tanhf(inn + r * hn);
  float hnew = (1.0f - zg) * nn + zg * hp;
  hout[(size_t)b * 256 + h] = hnew;
  if (hout16) hout16[(size_t)b * 256 + h] = (half_t)hnew;
  if (yout)   yout[(size_t)b * y_ld + h]  = (half_t)hnew;
}

__global__ void k_readout(const float* __restrict__ a, const float* __restrict__ b,
                          const float* __restrict__ c, const float* __restrict__ d,
                          float* __restrict__ out) {
  size_t i = (size_t)blockIdx.x * 256 + threadIdx.x;
  out[i] = a[i] + b[i] + c[i] + d[i];
}

// ------------------------------------------------------------------- driver

extern "C" void kernel_launch(void* const* d_in, const int* in_sizes, int n_in,
                              void* d_out, int out_size, void* d_ws, size_t ws_size,
                              hipStream_t stream) {
  const int N = 100000, E = 200000, B = 4096, H = 256;

  const int*   wid   = (const int*)d_in[0];
  const int*   src   = (const int*)d_in[1];
  const int*   dst   = (const int*)d_in[2];
  const int*   n2g   = (const int*)d_in[3];
  const float* emb   = (const float*)d_in[4];
  const float* gat_W = (const float*)d_in[5];
  const float* gat_as = (const float*)d_in[6];
  const float* gat_ad = (const float*)d_in[7];
  const float* gat_ga = (const float*)d_in[8];
  const float* gat_be = (const float*)d_in[9];
  const float* W_out = (const float*)d_in[10];
  const float* b_out = (const float*)d_in[11];
  const float* Wih0  = (const float*)d_in[12];
  const float* Whh0  = (const float*)d_in[13];
  const float* bih0  = (const float*)d_in[14];
  const float* bhh0  = (const float*)d_in[15];
  const float* Wih1  = (const float*)d_in[16];
  const float* Whh1  = (const float*)d_in[17];
  const float* bih1  = (const float*)d_in[18];
  const float* bhh1  = (const float*)d_in[19];

  float* out_newh = (float*)d_out;                 // [N,H]
  float* out_read = out_newh + (size_t)N * H;      // [B,H]

  // ---- workspace carve-up (bump allocator, 256B aligned) ----
  char* wp = (char*)d_ws;
  auto alloc = [&](size_t bytes) -> void* {
    void* p = (void*)wp;
    wp += (bytes + 255) & ~(size_t)255;
    return p;
  };
  half_t*   h0h    = (half_t*)alloc((size_t)N * H * 2);
  float*    z      = (float*) alloc((size_t)N * H * 4);
  float*    hagg   = (float*) alloc((size_t)N * H * 4);
  half_t*   catH   = (half_t*)alloc((size_t)N * 4 * H * 2);
  float*    mean1  = (float*) alloc((size_t)B * H * 4);
  float*    mean2  = (float*) alloc((size_t)B * H * 4);
  float*    cnt    = (float*) alloc((size_t)B * 4);
  float*    ss     = (float*) alloc((size_t)N * 4);
  float*    sd     = (float*) alloc((size_t)N * 4);
  unsigned* mU     = (unsigned*)alloc((size_t)N * 4);
  float*    ssum   = (float*) alloc((size_t)N * 4);
  float*    exb    = (float*) alloc((size_t)E * 4);
  float*    csum   = (float*) alloc((size_t)H * 4);
  float*    csq    = (float*) alloc((size_t)H * 4);
  half_t*   gatWh  = (half_t*)alloc((size_t)4 * H * H * 2);
  half_t*   Wouth  = (half_t*)alloc((size_t)H * 4 * H * 2);
  half_t*   Wih0h  = (half_t*)alloc((size_t)2 * 768 * 256 * 2);
  half_t*   Whh0h  = (half_t*)alloc((size_t)2 * 768 * 256 * 2);
  half_t*   Wih1h  = (half_t*)alloc((size_t)2 * 768 * 512 * 2);
  half_t*   Whh1h  = (half_t*)alloc((size_t)2 * 768 * 256 * 2);
  half_t*   xh     = (half_t*)alloc((size_t)B * 2 * H * 2);
  float*    giBuf  = (float*) alloc((size_t)2 * B * 768 * 4);
  float*    ghBuf  = (float*) alloc((size_t)B * 768 * 4);
  half_t*   y0h    = (half_t*)alloc((size_t)B * 2 * 512 * 2);
  float*    hstep  = (float*) alloc((size_t)B * H * 4);
  half_t*   hstep16= (half_t*)alloc((size_t)B * H * 2);
  float*    hlast[4];
  for (int i = 0; i < 4; ++i) hlast[i] = (float*)alloc((size_t)B * H * 4);

  auto gemm = [&](const half_t* A, const half_t* Bw, float* C, const float* bias,
                  int M, int Nc, int K) {
    int tiles = (M / 32) * (Nc / 32);   // one wave per 32x32 block
    k_wmma_gemm_nt<<<(tiles + 7) / 8, 256, 0, stream>>>(A, Bw, C, bias, M, Nc, K);
  };
  auto conv = [&](const float* in, half_t* outp, int n) {
    k_f32_to_f16<<<(n + 255) / 256, 256, 0, stream>>>(in, outp, n);
  };

  // ---- weight conversion (f32 -> f16) ----
  conv(gat_W, gatWh, 4 * H * H);
  conv(W_out, Wouth, H * 4 * H);
  conv(Wih0,  Wih0h, 2 * 768 * 256);
  conv(Whh0,  Whh0h, 2 * 768 * 256);
  conv(Wih1,  Wih1h, 2 * 768 * 512);
  conv(Whh1,  Whh1h, 2 * 768 * 256);

  // ---- embedding + graph means prep ----
  hipMemsetAsync(mean1, 0, (size_t)B * H * 4, stream);
  hipMemsetAsync(mean2, 0, (size_t)B * H * 4, stream);
  hipMemsetAsync(cnt,   0, (size_t)B * 4, stream);
  k_embed<<<N, 256, 0, stream>>>(wid, n2g, emb, h0h, mean1, cnt);
  k_div_mean1<<<B, 256, 0, stream>>>(mean1, cnt);

  // ---- 4 GAT layers (all consume h0) ----
  for (int i = 0; i < 4; ++i) {
    hipMemsetAsync(mU,   0, (size_t)N * 4, stream);
    hipMemsetAsync(ssum, 0, (size_t)N * 4, stream);
    hipMemsetAsync(hagg, 0, (size_t)N * H * 4, stream);
    hipMemsetAsync(csum, 0, (size_t)H * 4, stream);
    hipMemsetAsync(csq,  0, (size_t)H * 4, stream);

    gemm(h0h, gatWh + (size_t)i * H * H, z, nullptr, N, H, H);   // z = h0 @ W^T
    k_node_scores<<<N, 256, 0, stream>>>(z, gat_as + i * H, gat_ad + i * H, ss, sd);
    k_edge_max<<<(E + 255) / 256, 256, 0, stream>>>(src, dst, ss, sd, mU, E);
    k_edge_exp<<<(E + 255) / 256, 256, 0, stream>>>(src, dst, ss, sd, mU, exb, ssum, E);
    k_edge_scatter<<<E, 256, 0, stream>>>(src, dst, exb, ssum, z, hagg);
    k_bn_stats<<<(N + 255) / 256, 256, 0, stream>>>(hagg, csum, csq, N);
    k_bn_norm<<<N, 256, 0, stream>>>(hagg, csum, csq, gat_ga + i * H, gat_be + i * H,
                                     catH, i, 1.0f / (float)N);
  }

  // ---- new_h = concat @ W_out^T + b_out  (straight into d_out) ----
  gemm(catH, Wouth, out_newh, b_out, N, H, 4 * H);
  k_mean2_scatter<<<N, 256, 0, stream>>>(n2g, out_newh, mean2);
  k_build_x<<<B, 256, 0, stream>>>(mean1, mean2, cnt, xh);

  // ---- GRU layer 0 (bidirectional), seq len 2 ----
  for (int d = 0; d < 2; ++d) {
    int ta = (d == 0) ? 0 : 1, tb = 1 - ta;
    gemm(xh, Wih0h + (size_t)d * 768 * 256, giBuf, bih0 + d * 768, 2 * B, 768, 256);
    // step A: h_prev = 0 -> gh = bhh (broadcast)
    k_gru_cell<<<B, 256, 0, stream>>>(giBuf + ta * 768, 2 * 768, bhh0 + d * 768, 0,
                                      nullptr, hstep, hstep16,
                                      y0h + ta * 512 + d * 256, 1024);
    gemm(hstep16, Whh0h + (size_t)d * 768 * 256, ghBuf, bhh0 + d * 768, B, 768, 256);
    k_gru_cell<<<B, 256, 0, stream>>>(giBuf + tb * 768, 2 * 768, ghBuf, 768,
                                      hstep, hlast[d], nullptr,
                                      y0h + tb * 512 + d * 256, 1024);
  }

  // ---- GRU layer 1 (input = y0 [B,2,512]) ----
  for (int d = 0; d < 2; ++d) {
    int ta = (d == 0) ? 0 : 1, tb = 1 - ta;
    gemm(y0h, Wih1h + (size_t)d * 768 * 512, giBuf, bih1 + d * 768, 2 * B, 768, 512);
    k_gru_cell<<<B, 256, 0, stream>>>(giBuf + ta * 768, 2 * 768, bhh1 + d * 768, 0,
                                      nullptr, hstep, hstep16, nullptr, 0);
    gemm(hstep16, Whh1h + (size_t)d * 768 * 256, ghBuf, bhh1 + d * 768, B, 768, 256);
    k_gru_cell<<<B, 256, 0, stream>>>(giBuf + tb * 768, 2 * 768, ghBuf, 768,
                                      hstep, hlast[2 + d], nullptr, nullptr, 0);
  }

  // ---- readout: sum of 4 directional last states ----
  k_readout<<<B, 256, 0, stream>>>(hlast[0], hlast[1], hlast[2], hlast[3], out_read);
}
